// LFQBert_85066122264795
// MI455X (gfx1250) — compile-verified, matrix operand order checked by reading, not run on hardware
//
#include <hip/hip_runtime.h>

// ---------------------------------------------------------------------------
// LFQ-BERT encoder for gfx1250 (MI455X): bf16 WMMA GEMMs with async
// global->LDS double-buffered staging + VALU aux kernels.
// ---------------------------------------------------------------------------

typedef __bf16 bf16_t;
typedef __attribute__((ext_vector_type(16))) __bf16 v16bf;
typedef __attribute__((ext_vector_type(8)))  float  v8f;

#define DEPTH 12
#define DMODEL 768
#define NHEAD 8
#define DHEAD 96
#define DMLP 3072
#define BATCH 16
#define SEQ 256
#define NTOK (BATCH * SEQ)   // 4096
#define NBITS 10

// float -> bf16 with round-to-nearest-even
static __device__ __forceinline__ bf16_t f2bf(float f) {
    union { float f; unsigned u; } in; in.f = f;
    unsigned r = (in.u + 0x7FFFu + ((in.u >> 16) & 1u)) >> 16;
    unsigned short us = (unsigned short)r;
    bf16_t b;
    __builtin_memcpy(&b, &us, 2);
    return b;
}

// async 16B global -> LDS copy (ASYNCcnt-tracked), per-lane addresses
static __device__ __forceinline__ void async_ld16(unsigned lds_off,
                                                  unsigned long long gaddr) {
    asm volatile("global_load_async_to_lds_b128 %0, %1, off"
                 :: "v"(lds_off), "v"(gaddr) : "memory");
}
static __device__ __forceinline__ void wait_async0() {
    asm volatile("s_wait_asynccnt 0x0" ::: "memory");
}

// ---------------------------------------------------------------------------
// GEMM: C[M,N] = A[M,K](bf16) * W[N,K]^T(bf16) + bias[N], f32 accumulate.
// Block tile 128x128, BK=32, 256 threads = 8 waves (2x4 wave grid).
// Each wave: 4 M-subtiles x 2 N-subtiles -> 8 v_wmma per K-step.
// Double-buffered LDS fed by global_load_async_to_lds_b128.
// ---------------------------------------------------------------------------
#define BM 128
#define BN 128
#define BK 32
#define SA 40   // padded LDS row stride in bf16 elems (80B rows, 16B aligned)

__global__ __launch_bounds__(256)
void gemm_bf16_wmma(const bf16_t* __restrict__ A,
                    const bf16_t* __restrict__ W,
                    const float*  __restrict__ bias,
                    float* __restrict__ C,
                    int M, int N, int K)
{
    __shared__ __align__(16) bf16_t sA[2][BM * SA];
    __shared__ __align__(16) bf16_t sB[2][BN * SA];

    const int tid  = threadIdx.x;
    const int wave = tid >> 5;          // 0..7
    const int lane = tid & 31;
    const int wm   = wave >> 2;         // 0..1 : 64-row band
    const int wn   = wave & 3;          // 0..3 : 32-col band
    const int r    = lane & 15;
    const int half = lane >> 4;
    const int bm = blockIdx.y * BM;
    const int bn = blockIdx.x * BN;

    // cooperative tile fill: 512 chunks of 16B per matrix, 2 per thread
    const int lr = tid >> 2;            // 0..63
    const int lc = (tid & 3) * 8;       // bf16 elem offset within 32-elem row

    const unsigned long long gA0 = (unsigned long long)(uintptr_t)(A + (size_t)(bm + lr) * K + lc);
    const unsigned long long gA1 = (unsigned long long)(uintptr_t)(A + (size_t)(bm + 64 + lr) * K + lc);
    const unsigned long long gB0 = (unsigned long long)(uintptr_t)(W + (size_t)(bn + lr) * K + lc);
    const unsigned long long gB1 = (unsigned long long)(uintptr_t)(W + (size_t)(bn + 64 + lr) * K + lc);

    unsigned lA0[2], lA1[2], lB0[2], lB1[2];
#pragma unroll
    for (int b = 0; b < 2; ++b) {
        lA0[b] = (unsigned)(uintptr_t)(void*)&sA[b][lr * SA + lc];
        lA1[b] = (unsigned)(uintptr_t)(void*)&sA[b][(64 + lr) * SA + lc];
        lB0[b] = (unsigned)(uintptr_t)(void*)&sB[b][lr * SA + lc];
        lB1[b] = (unsigned)(uintptr_t)(void*)&sB[b][(64 + lr) * SA + lc];
    }

    v8f acc[4][2];
#pragma unroll
    for (int i = 0; i < 4; ++i)
#pragma unroll
        for (int j = 0; j < 2; ++j)
            acc[i][j] = (v8f){0.f, 0.f, 0.f, 0.f, 0.f, 0.f, 0.f, 0.f};

    // prologue: stage tile 0 into buffer 0
    {
        const unsigned long long kb = 0;
        async_ld16(lA0[0], gA0 + kb);
        async_ld16(lA1[0], gA1 + kb);
        async_ld16(lB0[0], gB0 + kb);
        async_ld16(lB1[0], gB1 + kb);
    }
    wait_async0();
    __syncthreads();

    const int nk = K / BK;
    for (int ks = 0; ks < nk; ++ks) {
        const int cur = ks & 1;
        const int nxt = cur ^ 1;

        if (ks + 1 < nk) {
            const unsigned long long kb = (unsigned long long)(ks + 1) * BK * 2; // bytes
            async_ld16(lA0[nxt], gA0 + kb);
            async_ld16(lA1[nxt], gA1 + kb);
            async_ld16(lB0[nxt], gB0 + kb);
            async_ld16(lB1[nxt], gB1 + kb);
        }

        union Frag { v16bf v; uint4 q[2]; } af[4], bfr[2];
#pragma unroll
        for (int i = 0; i < 4; ++i) {
            const int arow = wm * 64 + i * 16 + r;
            af[i].q[0] = *(const uint4*)(&sA[cur][arow * SA + half * 8]);
            af[i].q[1] = *(const uint4*)(&sA[cur][arow * SA + 16 + half * 8]);
        }
#pragma unroll
        for (int j = 0; j < 2; ++j) {
            const int brow = wn * 32 + j * 16 + r;
            bfr[j].q[0] = *(const uint4*)(&sB[cur][brow * SA + half * 8]);
            bfr[j].q[1] = *(const uint4*)(&sB[cur][brow * SA + 16 + half * 8]);
        }

#pragma unroll
        for (int i = 0; i < 4; ++i)
#pragma unroll
            for (int j = 0; j < 2; ++j)
                acc[i][j] = __builtin_amdgcn_wmma_f32_16x16x32_bf16(
                    false, af[i].v, false, bfr[j].v, (short)0, acc[i][j], false, false);

        if (ks + 1 < nk) {
            wait_async0();       // next buffer fully in LDS
            __syncthreads();     // ...and all waves done reading current buffer
        }
    }

    // epilogue: C/D layout vgpr e -> row (e + 8*half), lane%16 -> column
#pragma unroll
    for (int i = 0; i < 4; ++i) {
        const int row0 = bm + wm * 64 + i * 16 + half * 8;
#pragma unroll
        for (int j = 0; j < 2; ++j) {
            const int col = bn + wn * 32 + j * 16 + r;
            const float b = bias[col];
#pragma unroll
            for (int e = 0; e < 8; ++e)
                C[(size_t)(row0 + e) * N + col] = acc[i][j][e] + b;
        }
    }
}

// ---------------------------------------------------------------------------
// Elementwise / reduction kernels
// ---------------------------------------------------------------------------
__global__ void cast_f32_bf16(const float* __restrict__ in,
                              bf16_t* __restrict__ out, size_t n)
{
    size_t i = (size_t)blockIdx.x * blockDim.x + threadIdx.x;
    if (i < n) out[i] = f2bf(in[i]);
}

__global__ void init_h(const float* __restrict__ x,
                       float* __restrict__ h, bf16_t* __restrict__ hb, size_t n)
{
    size_t i = (size_t)blockIdx.x * blockDim.x + threadIdx.x;
    if (i < n) { float v = x[i]; h[i] = v; hb[i] = f2bf(v); }
}

// scores + softmax: one block per (b,h,q); 256 threads = one key each
__global__ __launch_bounds__(256)
void attn_scores_softmax(const float* __restrict__ qkv,   // [NTOK, 3*DMODEL]
                         float* __restrict__ attnW)       // [B,H,L,L]
{
    const int b = blockIdx.z, h = blockIdx.y, q = blockIdx.x;
    const int t = threadIdx.x;
    __shared__ float qs[DHEAD];
    __shared__ float red[256];

    const float* qrow = qkv + (size_t)(b * SEQ + q) * (3 * DMODEL) + h * DHEAD;
    if (t < DHEAD) qs[t] = qrow[t];
    __syncthreads();

    const float* krow = qkv + (size_t)(b * SEQ + t) * (3 * DMODEL) + DMODEL + h * DHEAD;
    float s = 0.f;
#pragma unroll 4
    for (int d = 0; d < DHEAD; ++d) s += qs[d] * krow[d];
    s *= 0.10206207261596577f;  // 1/sqrt(96)

    red[t] = s; __syncthreads();
    for (int st = 128; st > 0; st >>= 1) {
        if (t < st) red[t] = fmaxf(red[t], red[t + st]);
        __syncthreads();
    }
    const float m = red[0]; __syncthreads();
    const float e = expf(s - m);
    red[t] = e; __syncthreads();
    for (int st = 128; st > 0; st >>= 1) {
        if (t < st) red[t] += red[t + st];
        __syncthreads();
    }
    const float inv = 1.f / red[0];
    attnW[((size_t)((b * NHEAD + h) * SEQ + q)) * SEQ + t] = e * inv;
}

// o = attn * V : one block per (b,h,q); 96 threads = one head dim each
__global__ __launch_bounds__(96)
void attn_out(const float* __restrict__ qkv,
              const float* __restrict__ attnW,
              bf16_t* __restrict__ ao_bf)                 // [NTOK, DMODEL]
{
    const int b = blockIdx.z, h = blockIdx.y, q = blockIdx.x;
    const int t = threadIdx.x;                            // 0..95
    __shared__ float aw[SEQ];
    const float* arow = attnW + ((size_t)((b * NHEAD + h) * SEQ + q)) * SEQ;
    for (int i = t; i < SEQ; i += 96) aw[i] = arow[i];
    __syncthreads();

    const float* vcol = qkv + (size_t)(b * SEQ) * (3 * DMODEL) + 2 * DMODEL + h * DHEAD + t;
    float o = 0.f;
#pragma unroll 4
    for (int k = 0; k < SEQ; ++k) o += aw[k] * vcol[(size_t)k * (3 * DMODEL)];
    ao_bf[(size_t)(b * SEQ + q) * DMODEL + h * DHEAD + t] = f2bf(o);
}

// h = LayerNorm(a + h) * g + b ; writes f32 + bf16 copies (in-place on h)
__global__ __launch_bounds__(256)
void add_ln(const float* __restrict__ a,
            float* __restrict__ h, bf16_t* __restrict__ hb,
            const float* __restrict__ g, const float* __restrict__ beta)
{
    const int tok = blockIdx.x;
    const int t = threadIdx.x;
    __shared__ float red[256];
    __shared__ float red2[256];

    const float* ar = a + (size_t)tok * DMODEL;
    float*       hr = h + (size_t)tok * DMODEL;
    float x[3], s = 0.f, s2 = 0.f;
#pragma unroll
    for (int i = 0; i < 3; ++i) {
        float v = ar[t + i * 256] + hr[t + i * 256];
        x[i] = v; s += v; s2 += v * v;
    }
    red[t] = s; red2[t] = s2; __syncthreads();
    for (int st = 128; st > 0; st >>= 1) {
        if (t < st) { red[t] += red[t + st]; red2[t] += red2[t + st]; }
        __syncthreads();
    }
    const float mean = red[0] * (1.f / DMODEL);
    const float var  = red2[0] * (1.f / DMODEL) - mean * mean;
    const float inv  = rsqrtf(var + 1e-12f);
#pragma unroll
    for (int i = 0; i < 3; ++i) {
        const int d = t + i * 256;
        const float y = (x[i] - mean) * inv * g[d] + beta[d];
        hr[d] = y;
        hb[(size_t)tok * DMODEL + d] = f2bf(y);
    }
}

// exact GELU, f32 -> bf16
__global__ void gelu_bf16(const float* __restrict__ in,
                          bf16_t* __restrict__ out, size_t n)
{
    size_t i = (size_t)blockIdx.x * blockDim.x + threadIdx.x;
    if (i < n) {
        float v = in[i];
        out[i] = f2bf(0.5f * v * (1.f + erff(v * 0.70710678118654752f)));
    }
}

// LFQ head: z = h*pw^T + pb ; quantized = sign(z); index = packed signs;
// per-token softmax entropy over 1024 implicit codes.
__global__ __launch_bounds__(256)
void lfq_head(const float* __restrict__ h,
              const float* __restrict__ pw, const float* __restrict__ pb,
              float* __restrict__ quant, float* __restrict__ idx,
              float* __restrict__ ent)
{
    const int tok = blockIdx.x;
    const int t = threadIdx.x;
    __shared__ float red[256];
    __shared__ float z[NBITS];

    const float* hr = h + (size_t)tok * DMODEL;
    for (int k = 0; k < NBITS; ++k) {
        float s = 0.f;
        for (int i = t; i < DMODEL; i += 256) s += hr[i] * pw[k * DMODEL + i];
        red[t] = s; __syncthreads();
        for (int st = 128; st > 0; st >>= 1) {
            if (t < st) red[t] += red[t + st];
            __syncthreads();
        }
        if (t == 0) z[k] = red[0] + pb[k];
        __syncthreads();
    }

    if (t < NBITS) quant[(size_t)tok * NBITS + t] = (z[t] > 0.f) ? 1.f : -1.f;
    if (t == 0) {
        int c = 0;
        for (int k = 0; k < NBITS; ++k) if (z[k] > 0.f) c |= (1 << k);
        idx[tok] = (float)c;
    }

    // entropy: 4 codes per thread
    float l[4], lmax = -3.4e38f;
#pragma unroll
    for (int j = 0; j < 4; ++j) {
        const int c = t * 4 + j;
        float s = 0.f;
#pragma unroll
        for (int k = 0; k < NBITS; ++k) s += ((c >> k) & 1) ? z[k] : -z[k];
        l[j] = 2.f * s;
        lmax = fmaxf(lmax, l[j]);
    }
    red[t] = lmax; __syncthreads();
    for (int st = 128; st > 0; st >>= 1) {
        if (t < st) red[t] = fmaxf(red[t], red[t + st]);
        __syncthreads();
    }
    const float m = red[0]; __syncthreads();

    float S = 0.f, T = 0.f;
#pragma unroll
    for (int j = 0; j < 4; ++j) { float e = expf(l[j] - m); S += e; T += l[j] * e; }
    red[t] = S; __syncthreads();
    for (int st = 128; st > 0; st >>= 1) {
        if (t < st) red[t] += red[t + st];
        __syncthreads();
    }
    const float Ssum = red[0]; __syncthreads();
    red[t] = T; __syncthreads();
    for (int st = 128; st > 0; st >>= 1) {
        if (t < st) red[t] += red[t + st];
        __syncthreads();
    }
    const float Tsum = red[0];
    if (t == 0) ent[tok] = (m + logf(Ssum)) - Tsum / Ssum;
}

__global__ __launch_bounds__(256)
void reduce_entropy(const float* __restrict__ ent, float* __restrict__ out)
{
    __shared__ float red[256];
    const int t = threadIdx.x;
    float s = 0.f;
    for (int i = t; i < NTOK; i += 256) s += ent[i];
    red[t] = s; __syncthreads();
    for (int st = 128; st > 0; st >>= 1) {
        if (t < st) red[t] += red[t + st];
        __syncthreads();
    }
    if (t == 0) out[0] = red[0] * (1.f / NTOK);
}

// ---------------------------------------------------------------------------
// Host launcher
// ---------------------------------------------------------------------------
extern "C" void kernel_launch(void* const* d_in, const int* in_sizes, int n_in,
                              void* d_out, int out_size, void* d_ws, size_t ws_size,
                              hipStream_t stream)
{
    (void)in_sizes; (void)n_in; (void)out_size; (void)ws_size;

    const float* x    = (const float*)d_in[0];
    const float* wqkv = (const float*)d_in[1];
    const float* bqkv = (const float*)d_in[2];
    const float* wo   = (const float*)d_in[3];
    const float* bo   = (const float*)d_in[4];
    const float* g1   = (const float*)d_in[5];
    const float* b1   = (const float*)d_in[6];
    const float* w1   = (const float*)d_in[7];
    const float* bf1  = (const float*)d_in[8];
    const float* w2   = (const float*)d_in[9];
    const float* bf2  = (const float*)d_in[10];
    const float* g2   = (const float*)d_in[11];
    const float* b2   = (const float*)d_in[12];
    const float* pw   = (const float*)d_in[13];
    const float* pb   = (const float*)d_in[14];

    float* out = (float*)d_out;
    float* quant_out = out;                          // [NTOK, NBITS]
    float* idx_out   = out + (size_t)NTOK * NBITS;   // [NTOK]
    float* ent_out   = idx_out + NTOK;               // [1]

    // workspace carve-up
    char* ws = (char*)d_ws;
    size_t off = 0;
    auto alloc = [&](size_t bytes) -> char* {
        char* p = ws + off;
        off += (bytes + 255) & ~(size_t)255;
        return p;
    };
    const size_t N_WQKV = (size_t)DEPTH * 3 * DMODEL * DMODEL;
    const size_t N_WO   = (size_t)DEPTH * DMODEL * DMODEL;
    const size_t N_W1   = (size_t)DEPTH * DMLP * DMODEL;
    const size_t N_W2   = (size_t)DEPTH * DMODEL * DMLP;

    bf16_t* wqkv_bf = (bf16_t*)alloc(N_WQKV * 2);
    bf16_t* wo_bf   = (bf16_t*)alloc(N_WO * 2);
    bf16_t* w1_bf   = (bf16_t*)alloc(N_W1 * 2);
    bf16_t* w2_bf   = (bf16_t*)alloc(N_W2 * 2);
    float*  hf32    = (float*) alloc((size_t)NTOK * DMODEL * 4);
    bf16_t* hbf     = (bf16_t*)alloc((size_t)NTOK * DMODEL * 2);
    float*  big     = (float*) alloc((size_t)NTOK * DMLP * 4);   // qkv (2304) / ffn1 (3072)
    float*  attnW   = (float*) alloc((size_t)BATCH * NHEAD * SEQ * SEQ * 4);
    bf16_t* ao_bf   = (bf16_t*)alloc((size_t)NTOK * DMODEL * 2);
    float*  tmp768  = (float*) alloc((size_t)NTOK * DMODEL * 4); // proj / ffn2 out
    bf16_t* ffn1_bf = (bf16_t*)alloc((size_t)NTOK * DMLP * 2);
    float*  entbuf  = (float*) alloc((size_t)NTOK * 4);

    auto cast = [&](const float* src, bf16_t* dst, size_t n) {
        cast_f32_bf16<<<(unsigned)((n + 255) / 256), 256, 0, stream>>>(src, dst, n);
    };
    cast(wqkv, wqkv_bf, N_WQKV);
    cast(wo,   wo_bf,   N_WO);
    cast(w1,   w1_bf,   N_W1);
    cast(w2,   w2_bf,   N_W2);

    {
        const size_t n = (size_t)NTOK * DMODEL;
        init_h<<<(unsigned)((n + 255) / 256), 256, 0, stream>>>(x, hf32, hbf, n);
    }

    const dim3 attnGrid(SEQ, NHEAD, BATCH);
    for (int l = 0; l < DEPTH; ++l) {
        const bf16_t* wqkv_l = wqkv_bf + (size_t)l * 3 * DMODEL * DMODEL;
        const bf16_t* wo_l   = wo_bf   + (size_t)l * DMODEL * DMODEL;
        const bf16_t* w1_l   = w1_bf   + (size_t)l * DMLP * DMODEL;
        const bf16_t* w2_l   = w2_bf   + (size_t)l * DMODEL * DMLP;
        const float*  bqkv_l = bqkv + (size_t)l * 3 * DMODEL;
        const float*  bo_l   = bo   + (size_t)l * DMODEL;
        const float*  g1_l   = g1   + (size_t)l * DMODEL;
        const float*  b1_l   = b1   + (size_t)l * DMODEL;
        const float*  bf1_l  = bf1  + (size_t)l * DMLP;
        const float*  bf2_l  = bf2  + (size_t)l * DMODEL;
        const float*  g2_l   = g2   + (size_t)l * DMODEL;
        const float*  b2_l   = b2   + (size_t)l * DMODEL;

        // QKV GEMM: [4096,768] x [2304,768]^T
        gemm_bf16_wmma<<<dim3((3 * DMODEL) / BN, NTOK / BM), 256, 0, stream>>>(
            hbf, wqkv_l, bqkv_l, big, NTOK, 3 * DMODEL, DMODEL);

        attn_scores_softmax<<<attnGrid, 256, 0, stream>>>(big, attnW);
        attn_out<<<attnGrid, 96, 0, stream>>>(big, attnW, ao_bf);

        // WO GEMM: [4096,768] x [768,768]^T
        gemm_bf16_wmma<<<dim3(DMODEL / BN, NTOK / BM), 256, 0, stream>>>(
            ao_bf, wo_l, bo_l, tmp768, NTOK, DMODEL, DMODEL);

        add_ln<<<NTOK, 256, 0, stream>>>(tmp768, hf32, hbf, g1_l, b1_l);

        // W1 GEMM: [4096,768] x [3072,768]^T
        gemm_bf16_wmma<<<dim3(DMLP / BN, NTOK / BM), 256, 0, stream>>>(
            hbf, w1_l, bf1_l, big, NTOK, DMLP, DMODEL);

        {
            const size_t n = (size_t)NTOK * DMLP;
            gelu_bf16<<<(unsigned)((n + 255) / 256), 256, 0, stream>>>(big, ffn1_bf, n);
        }

        // W2 GEMM: [4096,3072] x [768,3072]^T
        gemm_bf16_wmma<<<dim3(DMODEL / BN, NTOK / BM), 256, 0, stream>>>(
            ffn1_bf, w2_l, bf2_l, tmp768, NTOK, DMODEL, DMLP);

        add_ln<<<NTOK, 256, 0, stream>>>(tmp768, hf32, hbf, g2_l, b2_l);
    }

    lfq_head<<<NTOK, 256, 0, stream>>>(hf32, pw, pb, quant_out, idx_out, entbuf);
    reduce_entropy<<<1, 256, 0, stream>>>(entbuf, ent_out);
}
